// SPINN_70411693850835
// MI455X (gfx1250) — matrix operand order repaired and checked by reading, not compiled, not run on hardware
//
#include <hip/hip_runtime.h>
#include <hip/hip_bf16.h>
#include <math.h>

// ---------------- problem constants ----------------
#define BB   128
#define LL   128
#define EE   512
#define HH   512
#define TT   255          // 2*L - 1
#define SS   130          // stack capacity L + 2
#define TWOH 1024
#define FIVEH 2560
#define ML   16384        // B * L rows of embedding GEMM

typedef __attribute__((ext_vector_type(16))) __bf16 v16bf;
typedef __attribute__((ext_vector_type(8)))  __bf16 v8bf;
typedef __attribute__((ext_vector_type(8)))  float  v8f;

__device__ __forceinline__ int iclamp(int v, int lo, int hi) {
    return v < lo ? lo : (v > hi ? hi : v);
}
__device__ __forceinline__ float sigf(float x) {
    return 1.0f / (1.0f + __expf(-x));
}

// ---------------- fp32 -> bf16 convert ----------------
__global__ __launch_bounds__(256) void cvt_bf16_kernel(const float* __restrict__ src,
                                                       __bf16* __restrict__ dst, int n) {
    for (int i = blockIdx.x * 256 + threadIdx.x; i < n; i += gridDim.x * 256)
        dst[i] = (__bf16)src[i];
}

// ---------------- WMMA bf16 GEMM: C[M,N] = A[M,K] @ B[K,N] + bias ----------------
// block: 256 threads (8 waves), tile 128(M) x 128(N), K-step 32, double-buffered LDS.
// wave (wm,wn) = (wave>>1, wave&1) owns a 32x64 patch = 2x4 wmma 16x16 tiles.
__global__ __launch_bounds__(256) void wmma_gemm_bf16_kernel(
    const __bf16* __restrict__ A, const __bf16* __restrict__ Bm,
    const float* __restrict__ bias, float* __restrict__ C,
    int M, int N, int K)
{
    __shared__ alignas(16) __bf16 As[2][128 * 32];   // 2 x 8 KB, row-major [m][k]
    __shared__ alignas(16) __bf16 Bt[2][128 * 32];   // 2 x 8 KB, transposed  [n][k]

    const int tid  = threadIdx.x;
    const int lane = tid & 31;
    const int wave = tid >> 5;
    const int lm   = lane & 15;     // lane 0..15 within half
    const int lh   = lane >> 4;     // lane-half 0/1
    const int nb   = N >> 7;        // N / 128 tiles
    const int bm   = (int)blockIdx.x / nb;
    const int bn   = (int)blockIdx.x % nb;
    const int m0   = bm << 7;
    const int n0   = bn << 7;
    const int wm   = wave >> 1;     // 0..3  (M sub-tile, 32 rows)
    const int wn   = wave & 1;      // 0..1  (N sub-tile, 64 cols)

    v8f acc[2][4];
    #pragma unroll
    for (int i = 0; i < 2; ++i)
        #pragma unroll
        for (int j = 0; j < 4; ++j)
            #pragma unroll
            for (int e = 0; e < 8; ++e) acc[i][j][e] = 0.0f;

    const int ar  = tid >> 1;            // A stage: row 0..127
    const int ac  = (tid & 1) << 4;      //          half-row chunk 0 / 16
    const int bk  = tid >> 3;            // B stage: k 0..31
    const int bg0 = (tid & 7) << 3;      //          first n group of 8

    // ---- stage buffer 0 (kk = 0) ----
    {
        const __bf16* ga = A + (size_t)(m0 + ar) * K + ac;
        *(v8bf*)(&As[0][ar * 32 + ac])     = *(const v8bf*)(ga);
        *(v8bf*)(&As[0][ar * 32 + ac + 8]) = *(const v8bf*)(ga + 8);
        #pragma unroll
        for (int h = 0; h < 2; ++h) {
            const int ng = bg0 + h * 64;
            const __bf16* gb = Bm + (size_t)bk * N + n0 + ng;
            v8bf bv = *(const v8bf*)gb;
            #pragma unroll
            for (int i = 0; i < 8; ++i) Bt[0][(ng + i) * 32 + bk] = bv[i];
        }
    }
    __syncthreads();

    const int nIter = K >> 5;
    for (int it = 0; it < nIter; ++it) {
        const int cur = it & 1;

        // ---- stage next K-slice into the other buffer (overlaps wmma below) ----
        if (it + 1 < nIter) {
            const int nxt = cur ^ 1;
            const int kk  = (it + 1) << 5;
            const __bf16* ga = A + (size_t)(m0 + ar) * K + kk + ac;
            *(v8bf*)(&As[nxt][ar * 32 + ac])     = *(const v8bf*)(ga);
            *(v8bf*)(&As[nxt][ar * 32 + ac + 8]) = *(const v8bf*)(ga + 8);
            #pragma unroll
            for (int h = 0; h < 2; ++h) {
                const int ng = bg0 + h * 64;
                const __bf16* gb = Bm + (size_t)(kk + bk) * N + n0 + ng;
                v8bf bv = *(const v8bf*)gb;
                #pragma unroll
                for (int i = 0; i < 8; ++i) Bt[nxt][(ng + i) * 32 + bk] = bv[i];
            }
            if (it + 2 < nIter) {   // prefetch the K-slice after that into cache
                const int kp = (it + 2) << 5;
                __builtin_prefetch(A + (size_t)(m0 + ar) * K + kp + ac, 0, 1);
                __builtin_prefetch(Bm + (size_t)(kp + bk) * N + n0 + bg0, 0, 1);
            }
        }

        // ---- load all fragments for this K-slice ----
        v16bf afr[2], bfr[4];
        #pragma unroll
        for (int mt = 0; mt < 2; ++mt) {
            // A fragment: lane-half lh holds K runs [lh*8, lh*8+7] and [16+lh*8 ...]
            const __bf16* ap = &As[cur][(wm * 32 + mt * 16 + lm) * 32 + lh * 8];
            v8bf alo = *(const v8bf*)ap;
            v8bf ahi = *(const v8bf*)(ap + 16);
            afr[mt] = __builtin_shufflevector(alo, ahi,
                0,1,2,3,4,5,6,7,8,9,10,11,12,13,14,15);
        }
        #pragma unroll
        for (int nt = 0; nt < 4; ++nt) {
            // B fragment: lane-half lh holds K runs [lh*8, lh*8+7] and [16+lh*8 ...]
            const __bf16* bp = &Bt[cur][(wn * 64 + nt * 16 + lm) * 32 + lh * 8];
            v8bf blo = *(const v8bf*)bp;
            v8bf bhi = *(const v8bf*)(bp + 16);
            bfr[nt] = __builtin_shufflevector(blo, bhi,
                0,1,2,3,4,5,6,7,8,9,10,11,12,13,14,15);
        }

        // ---- 2x4 WMMA tiles per wave ----
        #pragma unroll
        for (int mt = 0; mt < 2; ++mt)
            #pragma unroll
            for (int nt = 0; nt < 4; ++nt)
                acc[mt][nt] = __builtin_amdgcn_wmma_f32_16x16x32_bf16(
                    false, afr[mt], false, bfr[nt], (short)0, acc[mt][nt], false, false);

        __syncthreads();
    }

    // ---- epilogue: D layout -> VGPR r: M = lh*8 + r, N = lm ----
    #pragma unroll
    for (int mt = 0; mt < 2; ++mt)
        #pragma unroll
        for (int nt = 0; nt < 4; ++nt) {
            const int row0 = m0 + wm * 32 + mt * 16 + lh * 8;
            const int col  = n0 + wn * 64 + nt * 16 + lm;
            const float bvl = bias ? bias[col] : 0.0f;
            #pragma unroll
            for (int r = 0; r < 8; ++r)
                C[(size_t)(row0 + r) * N + col] = acc[mt][nt][r] + bvl;
        }
}

// ---------------- BatchNorm: deterministic two-pass stats ----------------
// Pass 1: 64 blocks, each reduces 256 rows; thread t owns channels t, t+256, t+512, t+768.
__global__ __launch_bounds__(256) void bn_partial_kernel(const float* __restrict__ out1,
                                                         float* __restrict__ part) {
    const int t = threadIdx.x;
    float s[4] = {0.f, 0.f, 0.f, 0.f};
    float q[4] = {0.f, 0.f, 0.f, 0.f};
    const int row0 = blockIdx.x * 256;
    for (int rr = 0; rr < 256; ++rr) {
        const float* rp = out1 + (size_t)(row0 + rr) * TWOH;
        #pragma unroll
        for (int c4 = 0; c4 < 4; ++c4) {
            float v = rp[t + c4 * 256];
            s[c4] += v;
            q[c4] += v * v;
        }
    }
    float* pb = part + (size_t)blockIdx.x * 2048;
    #pragma unroll
    for (int c4 = 0; c4 < 4; ++c4) {
        pb[c4 * 256 + t]        = s[c4];
        pb[1024 + c4 * 256 + t] = q[c4];
    }
}

// Pass 2: 1024 channels -> scale/shift
__global__ __launch_bounds__(256) void bn_final_kernel(const float* __restrict__ part,
                                                       const float* __restrict__ gamma,
                                                       const float* __restrict__ beta,
                                                       float* __restrict__ ss) {
    const int c = blockIdx.x * 256 + threadIdx.x;
    if (c >= TWOH) return;
    float s = 0.f, q = 0.f;
    for (int g = 0; g < 64; ++g) {
        s += part[(size_t)g * 2048 + c];
        q += part[(size_t)g * 2048 + 1024 + c];
    }
    const float inv_n = 1.0f / (float)ML;
    float mu  = s * inv_n;
    float var = q * inv_n - mu * mu;
    float sc  = gamma[c] * rsqrtf(var + 1e-5f);
    ss[c]        = sc;
    ss[TWOH + c] = beta[c] - mu * sc;
}

// apply BN in place (vectorized float4)
__global__ __launch_bounds__(256) void bn_apply_kernel(float* __restrict__ out1,
                                                       const float* __restrict__ ss) {
    const int total4 = (ML * TWOH) / 4;
    for (int i = blockIdx.x * 256 + threadIdx.x; i < total4; i += gridDim.x * 256) {
        float4 v = ((const float4*)out1)[i];
        int c0 = (i * 4) & (TWOH - 1);
        v.x = v.x * ss[c0 + 0] + ss[TWOH + c0 + 0];
        v.y = v.y * ss[c0 + 1] + ss[TWOH + c0 + 1];
        v.z = v.z * ss[c0 + 2] + ss[TWOH + c0 + 2];
        v.w = v.w * ss[c0 + 3] + ss[TWOH + c0 + 3];
        ((float4*)out1)[i] = v;
    }
}

// ---------------- scan step kernels ----------------
// gather: Astep[b] = bf16([ h_left | h_right ]) from stack
__global__ __launch_bounds__(256) void gather_kernel(const float* __restrict__ sh,
                                                     const int* __restrict__ ptr_cur,
                                                     __bf16* __restrict__ Astep) {
    const int b  = blockIdx.x;
    const int p  = ptr_cur[b];
    const int i1 = iclamp(p - 1, 0, SS - 1);
    const int i2 = iclamp(p - 2, 0, SS - 1);
    const float* hl = sh + ((size_t)b * SS + i2) * HH;
    const float* hr = sh + ((size_t)b * SS + i1) * HH;
    __bf16* ao = Astep + (size_t)b * TWOH;
    for (int j = threadIdx.x; j < HH; j += 256) {
        ao[j]      = (__bf16)hl[j];
        ao[HH + j] = (__bf16)hr[j];
    }
}

// update: gates -> TreeLSTM cell, select shift/reduce, write stack, advance ptr/bp
__global__ __launch_bounds__(256) void update_kernel(
    const float* __restrict__ a, const float* __restrict__ out1,
    float* __restrict__ sh, float* __restrict__ sc,
    const int* __restrict__ trans, int t,
    const int* __restrict__ ptr_c, const int* __restrict__ bp_c,
    int* __restrict__ ptr_n, int* __restrict__ bp_n)
{
    const int b   = blockIdx.x;
    const int act = trans[b * TT + t];
    const int p   = ptr_c[b];
    const int bp  = bp_c[b];
    const int i1  = iclamp(p - 1, 0, SS - 1);
    const int i2  = iclamp(p - 2, 0, SS - 1);
    const bool is_s = (act == 2);
    const bool is_r = (act == 3);
    const int pos = is_s ? iclamp(p, 0, SS - 1) : i2;
    const int bi  = iclamp(bp, 0, LL - 1);
    const bool wr = is_s || is_r;

    const float* ab = a + (size_t)b * FIVEH;
    const float* wb = out1 + ((size_t)b * LL + bi) * TWOH;

    for (int j = threadIdx.x; j < HH; j += 256) {
        float ig = ab[0 * HH + j];
        float fl = ab[1 * HH + j];
        float fr = ab[2 * HH + j];
        float og = ab[3 * HH + j];
        float gg = ab[4 * HH + j];
        float cl = sc[((size_t)b * SS + i2) * HH + j];
        float cr = sc[((size_t)b * SS + i1) * HH + j];
        float rc = sigf(fl) * cl + sigf(fr) * cr + sigf(ig) * tanhf(gg);
        float rh = sigf(og) * tanhf(rc);
        float wh = wb[j];
        float wc = wb[HH + j];
        float nh = is_s ? wh : rh;
        float nc = is_s ? wc : rc;
        size_t si = ((size_t)b * SS + pos) * HH + j;
        float vh = wr ? nh : sh[si];
        float vc = wr ? nc : sc[si];
        sh[si] = vh;
        sc[si] = vc;
    }
    if (threadIdx.x == 0) {
        ptr_n[b] = p + (is_s ? 1 : (is_r ? -1 : 0));
        bp_n[b]  = bp + (is_s ? 1 : 0);
    }
}

// final: out[b] = stack_h[b][ptr-1]
__global__ __launch_bounds__(256) void final_kernel(const float* __restrict__ sh,
                                                    const int* __restrict__ ptr_f,
                                                    float* __restrict__ out) {
    const int b = blockIdx.x;
    const int i = iclamp(ptr_f[b] - 1, 0, SS - 1);
    const float* src = sh + ((size_t)b * SS + i) * HH;
    for (int j = threadIdx.x; j < HH; j += 256)
        out[(size_t)b * HH + j] = src[j];
}

// ---------------- launch ----------------
static inline size_t alignup256(size_t x) { return (x + 255) & ~(size_t)255; }

extern "C" void kernel_launch(void* const* d_in, const int* in_sizes, int n_in,
                              void* d_out, int out_size, void* d_ws, size_t ws_size,
                              hipStream_t stream) {
    (void)in_sizes; (void)n_in; (void)out_size; (void)ws_size;
    const float* sentence = (const float*)d_in[0];   // [B,L,E]
    const int*   trans    = (const int*)d_in[1];     // [B,T] int32
    const float* word_W   = (const float*)d_in[2];   // [E,2H]
    const float* word_b   = (const float*)d_in[3];   // [2H]
    const float* gamma    = (const float*)d_in[4];   // [2H]
    const float* beta     = (const float*)d_in[5];   // [2H]
    const float* reduce_W = (const float*)d_in[6];   // [2H,5H]
    const float* reduce_b = (const float*)d_in[7];   // [5H]
    float* out = (float*)d_out;                      // [B,H]

    char* ws = (char*)d_ws;
    size_t off = 0;
    auto take = [&](size_t bytes) { size_t r = off; off += alignup256(bytes); return r; };
    size_t o_sentbf = take((size_t)BB * LL * EE * 2);       // sentence bf16
    size_t o_wWbf   = take((size_t)EE * TWOH * 2);          // word_W bf16
    size_t o_rWbf   = take((size_t)TWOH * FIVEH * 2);       // reduce_W bf16
    size_t o_out1   = take((size_t)ML * TWOH * 4);          // embedding output (BN'd in place)
    size_t o_bnp    = take((size_t)64 * 2048 * 4);          // BN partials
    size_t o_bnss   = take((size_t)2048 * 4);               // BN scale/shift
    size_t o_sh     = take((size_t)BB * SS * HH * 4);       // stack h
    size_t o_sc     = take((size_t)BB * SS * HH * 4);       // stack c
    size_t o_ast    = take((size_t)BB * TWOH * 2);          // step A (bf16)
    size_t o_astp   = take((size_t)BB * FIVEH * 4);         // step gates (fp32)
    size_t o_pb     = take((size_t)512 * 4);                // ptr/bp double buffers

    __bf16* sent_bf = (__bf16*)(ws + o_sentbf);
    __bf16* wW_bf   = (__bf16*)(ws + o_wWbf);
    __bf16* rW_bf   = (__bf16*)(ws + o_rWbf);
    float*  out1    = (float*)(ws + o_out1);
    float*  bn_part = (float*)(ws + o_bnp);
    float*  bn_ss   = (float*)(ws + o_bnss);
    float*  stack_h = (float*)(ws + o_sh);
    float*  stack_c = (float*)(ws + o_sc);
    __bf16* Astep   = (__bf16*)(ws + o_ast);
    float*  astep   = (float*)(ws + o_astp);
    int*    pb      = (int*)(ws + o_pb);
    // layout: ptr(buf) = pb + buf*256, bp(buf) = pb + buf*256 + 128

    // ---- precision conversion (one-time per launch) ----
    cvt_bf16_kernel<<<2048, 256, 0, stream>>>(sentence, sent_bf, BB * LL * EE);
    cvt_bf16_kernel<<<2048, 256, 0, stream>>>(word_W, wW_bf, EE * TWOH);
    cvt_bf16_kernel<<<2048, 256, 0, stream>>>(reduce_W, rW_bf, TWOH * FIVEH);

    // ---- embedding GEMM: [16384,512] x [512,1024] + word_b ----
    wmma_gemm_bf16_kernel<<<(ML / 128) * (TWOH / 128), 256, 0, stream>>>(
        sent_bf, wW_bf, word_b, out1, ML, TWOH, EE);

    // ---- BatchNorm (deterministic) ----
    bn_partial_kernel<<<64, 256, 0, stream>>>(out1, bn_part);
    bn_final_kernel<<<4, 256, 0, stream>>>(bn_part, gamma, beta, bn_ss);
    bn_apply_kernel<<<4096, 256, 0, stream>>>(out1, bn_ss);

    // ---- init scan state ----
    hipMemsetAsync(stack_h, 0, (size_t)BB * SS * HH * 4, stream);
    hipMemsetAsync(stack_c, 0, (size_t)BB * SS * HH * 4, stream);
    hipMemsetAsync(pb, 0, 512 * 4, stream);

    // ---- 255-step shift-reduce scan ----
    for (int t = 0; t < TT; ++t) {
        const int cur = t & 1, nxt = cur ^ 1;
        int* ptr_c = pb + cur * 256;
        int* bp_c  = pb + cur * 256 + 128;
        int* ptr_n = pb + nxt * 256;
        int* bp_n  = pb + nxt * 256 + 128;

        gather_kernel<<<BB, 256, 0, stream>>>(stack_h, ptr_c, Astep);
        wmma_gemm_bf16_kernel<<<(BB / 128) * (FIVEH / 128), 256, 0, stream>>>(
            Astep, rW_bf, reduce_b, astep, BB, FIVEH, TWOH);
        update_kernel<<<BB, 256, 0, stream>>>(astep, out1, stack_h, stack_c,
                                              trans, t, ptr_c, bp_c, ptr_n, bp_n);
    }

    // final state sits in buffer (TT & 1) == 1
    final_kernel<<<BB, 256, 0, stream>>>(stack_h, pb + (TT & 1) * 256, out);
}
